// SphereSelfAttention_70368744178015
// MI455X (gfx1250) — matrix operand order repaired and checked by reading, not compiled, not run on hardware
//
#include <hip/hip_runtime.h>
#include <math.h>

typedef __bf16 bf16;
typedef __bf16 v16bf __attribute__((ext_vector_type(16)));
typedef __bf16 v8bf  __attribute__((ext_vector_type(8)));
typedef float  v8f   __attribute__((ext_vector_type(8)));
typedef unsigned int u32x4 __attribute__((ext_vector_type(4)));
typedef int          i32x8 __attribute__((ext_vector_type(8)));
typedef int          i32x4 __attribute__((ext_vector_type(4)));

#define B_  2
#define H_  96
#define W_  96
#define C_  256
#define NH_ 8
#define DH_ 32
#define L_  2
#define FF_ 1024
#define N_  (H_*W_)        /* 9216  */
#define T_  (B_*N_*9)      /* 165888 tokens */

#if defined(__gfx1250__) && __has_builtin(__builtin_amdgcn_tensor_load_to_lds) && \
    __has_builtin(__builtin_amdgcn_s_wait_tensorcnt)
#define USE_TDM 1
#else
#define USE_TDM 0
#endif

// ---------------- f32 -> bf16 convert ----------------
__global__ void k_cvt_bf16(const float* __restrict__ src, bf16* __restrict__ dst, int n) {
    int i = blockIdx.x * blockDim.x + threadIdx.x;
    if (i < n) dst[i] = (bf16)src[i];
}

// ---------------- WMMA bf16 GEMM ----------------
// D[M,N] = A[M,K](bf16) x Wt[K,N](bf16) + bias, fused epilogue.
// block = 256 threads (8 wave32), tile BM=128 BN=64 BK=32.
// waves: 4 along M x 2 along N; each wave 32x32 via 2x2 fragments of 16x16.
// A tile staged via Tensor Data Mover (2D strided tile -> LDS), B tile via
// coalesced b128 global loads + transposed LDS stores.
// EPI: 0 = f32 out, 1 = f32 out = resid + acc, 2 = gelu->bf16, 3 = bf16 out
template<int EPI>
__global__ __launch_bounds__(256)
void k_gemm(const bf16* __restrict__ A, const bf16* __restrict__ Wt,
            const float* __restrict__ bias,
            float* __restrict__ outF, bf16* __restrict__ outB,
            const float* __restrict__ resid,
            int M, int N, int K)
{
    __shared__ __align__(16) bf16 As[128 * 32];   // [m][k] row-major (TDM dest)
    __shared__ __align__(16) bf16 Bs[64 * 32];    // transposed: [n][k]

    const int t    = threadIdx.x;
    const int lane = t & 31;
    const int wid  = t >> 5;
    const int wm   = wid & 3;
    const int wn   = wid >> 2;
    const int m0   = blockIdx.y * 128;
    const int n0   = blockIdx.x * 64;

    v8f acc[2][2];
    #pragma unroll
    for (int a = 0; a < 2; ++a)
        #pragma unroll
        for (int b = 0; b < 2; ++b) acc[a][b] = {};

#if !USE_TDM
    const int arow = t >> 1;            // 0..127
    const int acol = (t & 1) * 16;      // 0 / 16
#endif
    const int brow = t >> 3;            // 0..31  (k row)
    const int bcol = (t & 7) * 8;       // 0,8,..,56 (n col, 8 contiguous)

    for (int k0 = 0; k0 < K; k0 += 32) {
        if (k0 + 32 < K)                // gfx1250 global_prefetch_b8 of next B tile
            __builtin_prefetch(Wt + (size_t)(k0 + 32 + brow) * N + n0 + bcol, 0, 1);

#if USE_TDM
        // --- A tile via Tensor Data Mover: 128x32 bf16 tile, row stride K ---
        if (wid == 0) {
            unsigned long long ga = (unsigned long long)(uintptr_t)(A + (size_t)m0 * K + k0);
            unsigned int la = (unsigned int)(uintptr_t)(void*)As;   // LDS offset (low 32 bits)
            u32x4 g0;
            g0[0] = 1u;                                   // count=1, user mode, no gather
            g0[1] = la;                                   // lds_addr
            g0[2] = (unsigned int)ga;                     // global_addr[31:0]
            g0[3] = (unsigned int)((ga >> 32) & 0x01FFFFFFu) | (2u << 30); // addr[56:32], type=2
            i32x8 g1;
            g1[0] = (1 << 16);                            // wg_mask=0, data_size=1 (2 bytes)
            g1[1] = (32 & 0xFFFF) << 16;                  // tensor_dim0 = 32 (lo16 @ bits 63:48)
            g1[2] = (128 << 16);                          // tensor_dim0 hi=0 | tensor_dim1 = 128
            g1[3] = (32 << 16);                           // tensor_dim1 hi=0 | tile_dim0 = 32
            g1[4] = 128;                                  // tile_dim1 = 128, tile_dim2 = 0
            g1[5] = K;                                    // tensor_dim0_stride[31:0] = K
            g1[6] = 0;                                    // stride0 hi | stride1 lo
            g1[7] = 0;                                    // stride1 hi
            i32x4 gz4 = (i32x4)0;                         // groups 2/3: dims 2..4 unused
            i32x8 gz8 = (i32x8)0;                         // extra descriptor group (unused)
            __builtin_amdgcn_tensor_load_to_lds(g0, g1, gz4, gz4, gz8, 0);
            __builtin_amdgcn_s_wait_tensorcnt(0);
        }
#else
        // fallback: manual staging of A tile (row-major [128][32])
        {
            const bf16* ag = A + (size_t)(m0 + arow) * K + k0 + acol;
            *(v8bf*)&As[arow * 32 + acol]     = *(const v8bf*)(ag);
            *(v8bf*)&As[arow * 32 + acol + 8] = *(const v8bf*)(ag + 8);
        }
#endif
        // --- B tile: coalesced b128 global load, transpose on LDS store ---
        {
            v8bf bvec = *(const v8bf*)(Wt + (size_t)(k0 + brow) * N + n0 + bcol);
            #pragma unroll
            for (int j = 0; j < 8; ++j)
                Bs[(bcol + j) * 32 + brow] = bvec[j];
        }
        __syncthreads();

        union V16 { v16bf v; v8bf h[2]; };
        V16 af[2], bfr[2];
        // A fragment per ISA layout: lanes<16 -> K {0..7,16..23}, lanes>=16 -> +8
        const int khalf = (lane >> 4) * 8;
        #pragma unroll
        for (int fm = 0; fm < 2; ++fm) {
            int row = wm * 32 + fm * 16 + (lane & 15);
            af[fm].h[0] = *(const v8bf*)&As[row * 32 + khalf];
            af[fm].h[1] = *(const v8bf*)&As[row * 32 + 16 + khalf];
        }
        // B fragment: lane%16 = n, lanes<16 K {0..7,16..23}, lanes>=16 -> +8
        const int kbase = (lane >> 4) * 8;
        #pragma unroll
        for (int fn = 0; fn < 2; ++fn) {
            int col = wn * 32 + fn * 16 + (lane & 15);
            bfr[fn].h[0] = *(const v8bf*)&Bs[col * 32 + kbase];
            bfr[fn].h[1] = *(const v8bf*)&Bs[col * 32 + kbase + 16];
        }
        #pragma unroll
        for (int fm = 0; fm < 2; ++fm)
            #pragma unroll
            for (int fn = 0; fn < 2; ++fn)
                acc[fm][fn] = __builtin_amdgcn_wmma_f32_16x16x32_bf16(
                    false, af[fm].v, false, bfr[fn].v,
                    (short)0, acc[fm][fn], false, false);
        __syncthreads();
    }

    // epilogue: C/D layout — VGPR r: lanes 0-15 M=r, lanes 16-31 M=r+8; N=lane%16
    const int mrow_off = (lane >> 4) * 8;
    const int ncol0    = n0 + wn * 32 + (lane & 15);
    #pragma unroll
    for (int fm = 0; fm < 2; ++fm) {
        #pragma unroll
        for (int fn = 0; fn < 2; ++fn) {
            int nn = ncol0 + fn * 16;
            float bv = bias[nn];
            #pragma unroll
            for (int r = 0; r < 8; ++r) {
                int mm = m0 + wm * 32 + fm * 16 + mrow_off + r;
                size_t idx = (size_t)mm * N + nn;
                float v = acc[fm][fn][r] + bv;
                if (EPI == 0) {
                    outF[idx] = v;
                } else if (EPI == 1) {
                    outF[idx] = resid[idx] + v;
                } else if (EPI == 2) {
                    float x = v;
                    float g = 0.5f * x * (1.0f + tanhf(0.7978845608028654f *
                                     (x + 0.044715f * x * x * x)));
                    outB[idx] = (bf16)g;
                } else {
                    outB[idx] = (bf16)v;
                }
            }
        }
    }
}

// ---------------- grid_sample + positional encoding ----------------
// one block per (b,n,kk), one thread per channel c
__global__ void k_sample_pe(const float* __restrict__ xproj,   // [B,H,W,C] (== [B,N,C])
                            const float* __restrict__ sphere,  // [B,H,W,9,2]
                            const float* __restrict__ ppw,     // [2,C]
                            const float* __restrict__ ppb,     // [C]
                            float* __restrict__ xw,            // [T,C]
                            float* __restrict__ pew)           // [T,C]
{
    int e  = blockIdx.x;          // ((b*N + n)*9 + kk)
    int kk = e % 9;
    int bn = e / 9;
    int n  = bn % N_;
    int b  = bn / N_;
    int hi = n / W_;
    int wi = n % W_;

    size_t sp = ((((size_t)b * H_ + hi) * W_ + wi) * 9 + kk) * 2;
    float gxn = sphere[sp + 0];
    float gyn = sphere[sp + 1];
    float gx = (gxn + 1.0f) * (W_ * 0.5f) - 0.5f;
    float gy = (gyn + 1.0f) * (H_ * 0.5f) - 0.5f;
    float x0 = floorf(gx), y0 = floorf(gy);
    float wx1 = gx - x0, wx0 = 1.0f - wx1;
    float wy1 = gy - y0, wy0 = 1.0f - wy1;
    int ix0 = (int)x0, iy0 = (int)y0, ix1 = ix0 + 1, iy1 = iy0 + 1;
    float vx0 = (ix0 >= 0 && ix0 < W_) ? 1.f : 0.f;
    float vx1 = (ix1 >= 0 && ix1 < W_) ? 1.f : 0.f;
    float vy0 = (iy0 >= 0 && iy0 < H_) ? 1.f : 0.f;
    float vy1 = (iy1 >= 0 && iy1 < H_) ? 1.f : 0.f;
    int cx0 = min(max(ix0, 0), W_ - 1), cx1 = min(max(ix1, 0), W_ - 1);
    int cy0 = min(max(iy0, 0), H_ - 1), cy1 = min(max(iy1, 0), H_ - 1);
    size_t base = (size_t)b * N_ * C_;
    const float* p00 = xproj + base + ((size_t)cy0 * W_ + cx0) * C_;
    const float* p10 = xproj + base + ((size_t)cy0 * W_ + cx1) * C_;
    const float* p01 = xproj + base + ((size_t)cy1 * W_ + cx0) * C_;
    const float* p11 = xproj + base + ((size_t)cy1 * W_ + cx1) * C_;

    int c = threadIdx.x;
    float s = (wx0 * wy0) * vx0 * vy0 * p00[c]
            + (wx1 * wy0) * vx1 * vy0 * p10[c]
            + (wx0 * wy1) * vx0 * vy1 * p01[c]
            + (wx1 * wy1) * vx1 * vy1 * p11[c];
    float pe = gxn * ppw[c] + gyn * ppw[C_ + c] + ppb[c];
    size_t o = (size_t)e * C_ + c;
    xw[o]  = s + pe;
    pew[o] = pe;
}

// ---------------- layernorm (wave32 per row) ----------------
// MODE 0: out0 = bf16(h + pew), out1 = bf16(h)   (LN1 -> qk-input, v-input)
// MODE 1: out0 = bf16(h)                          (LN2 -> ffn input)
template<int MODE>
__global__ void k_ln(const float* __restrict__ xw, const float* __restrict__ pew,
                     const float* __restrict__ g, const float* __restrict__ bb,
                     bf16* __restrict__ out0, bf16* __restrict__ out1)
{
    int row  = blockIdx.x * 8 + (threadIdx.x >> 5);
    int lane = threadIdx.x & 31;
    const float* x = xw + (size_t)row * C_;
    float v[8], s = 0.f, s2 = 0.f;
    #pragma unroll
    for (int j = 0; j < 8; ++j) {
        v[j] = x[j * 32 + lane];
        s += v[j]; s2 += v[j] * v[j];
    }
    #pragma unroll
    for (int off = 16; off > 0; off >>= 1) {
        s  += __shfl_xor(s,  off, 32);
        s2 += __shfl_xor(s2, off, 32);
    }
    float mean = s * (1.0f / C_);
    float var  = s2 * (1.0f / C_) - mean * mean;
    float rstd = rsqrtf(var + 1e-5f);
    #pragma unroll
    for (int j = 0; j < 8; ++j) {
        int c = j * 32 + lane;
        float h = (v[j] - mean) * rstd * g[c] + bb[c];
        size_t o = (size_t)row * C_ + c;
        if (MODE == 0) { out0[o] = (bf16)(h + pew[o]); out1[o] = (bf16)h; }
        else           { out0[o] = (bf16)h; }
    }
}

// ---------------- 9x9 window attention ----------------
// one thread per (b,n,head,query i)
__global__ void k_attn(const bf16* __restrict__ q, const bf16* __restrict__ k,
                       const bf16* __restrict__ v, bf16* __restrict__ o)
{
    long gid = (long)blockIdx.x * blockDim.x + threadIdx.x;
    long total = (long)B_ * N_ * NH_ * 9;
    if (gid >= total) return;
    int i  = (int)(gid % 9);
    int hh = (int)((gid / 9) % NH_);
    long bn = gid / (9 * NH_);
    size_t base = (size_t)bn * 9 * C_ + (size_t)hh * DH_;

    float qr[DH_];
    const bf16* qp = q + base + (size_t)i * C_;
    #pragma unroll
    for (int d = 0; d < DH_; ++d) qr[d] = (float)qp[d];

    float sc[9], mx = -1e30f;
    for (int j = 0; j < 9; ++j) {
        const bf16* kp = k + base + (size_t)j * C_;
        float s = 0.f;
        #pragma unroll
        for (int d = 0; d < DH_; ++d) s += qr[d] * (float)kp[d];
        s *= 0.17677669529663687f;   // 1/sqrt(32)
        sc[j] = s;
        mx = fmaxf(mx, s);
    }
    float den = 0.f;
    for (int j = 0; j < 9; ++j) { sc[j] = expf(sc[j] - mx); den += sc[j]; }
    float inv = 1.0f / den;

    float ov[DH_];
    #pragma unroll
    for (int d = 0; d < DH_; ++d) ov[d] = 0.f;
    for (int j = 0; j < 9; ++j) {
        float w = sc[j] * inv;
        const bf16* vp = v + base + (size_t)j * C_;
        #pragma unroll
        for (int d = 0; d < DH_; ++d) ov[d] += w * (float)vp[d];
    }
    bf16* op = o + base + (size_t)i * C_;
    #pragma unroll
    for (int d = 0; d < DH_; ++d) op[d] = (bf16)ov[d];
}

// ---------------- final 3x3 re-interleave to [B, 3H, 3W, C] ----------------
__global__ void k_reorder(const float* __restrict__ xw, float* __restrict__ out)
{
    size_t gid = (size_t)blockIdx.x * blockDim.x + threadIdx.x;
    int c  = (int)(gid % C_);
    size_t r = gid / C_;
    int kk = (int)(r % 9); r /= 9;
    int n  = (int)(r % N_);
    int b  = (int)(r / N_);
    int hi = n / W_, wi = n % W_;
    int ki = kk / 3, kj = kk % 3;
    size_t dst = (((size_t)b * (3 * H_) + (3 * hi + ki)) * (3 * W_) + (3 * wi + kj)) * C_ + c;
    out[dst] = xw[gid];
}

extern "C" void kernel_launch(void* const* d_in, const int* in_sizes, int n_in,
                              void* d_out, int out_size, void* d_ws, size_t ws_size,
                              hipStream_t stream) {
    (void)in_sizes; (void)n_in; (void)out_size; (void)ws_size;

    const float* in_x   = (const float*)d_in[0];
    const float* sphere = (const float*)d_in[1];
    const float* ipw    = (const float*)d_in[2];
    const float* ipb    = (const float*)d_in[3];
    const float* ppw    = (const float*)d_in[4];
    const float* ppb    = (const float*)d_in[5];
    const float* wq = (const float*)d_in[6];  const float* bq = (const float*)d_in[7];
    const float* wk = (const float*)d_in[8];  const float* bk = (const float*)d_in[9];
    const float* wv = (const float*)d_in[10]; const float* bv = (const float*)d_in[11];
    const float* wo = (const float*)d_in[12]; const float* bo = (const float*)d_in[13];
    const float* ln1g = (const float*)d_in[14]; const float* ln1b = (const float*)d_in[15];
    const float* ln2g = (const float*)d_in[16]; const float* ln2b = (const float*)d_in[17];
    const float* w1 = (const float*)d_in[18]; const float* b1 = (const float*)d_in[19];
    const float* w2 = (const float*)d_in[20]; const float* b2 = (const float*)d_in[21];
    float* out = (float*)d_out;

    // ---- workspace bump allocator (256B aligned) ----
    char* p = (char*)d_ws;
    auto alloc = [&](size_t bytes) -> void* {
        void* r = (void*)p;
        p += (bytes + 255) & ~(size_t)255;
        return r;
    };
    const size_t TC = (size_t)T_ * C_;
    bf16*  inp_bf = (bf16*)alloc((size_t)B_ * N_ * C_ * 2);
    bf16*  ipw_bf = (bf16*)alloc((size_t)C_ * C_ * 2);
    bf16*  wq_bf  = (bf16*)alloc((size_t)L_ * C_ * C_ * 2);
    bf16*  wk_bf  = (bf16*)alloc((size_t)L_ * C_ * C_ * 2);
    bf16*  wv_bf  = (bf16*)alloc((size_t)L_ * C_ * C_ * 2);
    bf16*  wo_bf  = (bf16*)alloc((size_t)L_ * C_ * C_ * 2);
    bf16*  w1_bf  = (bf16*)alloc((size_t)L_ * C_ * FF_ * 2);
    bf16*  w2_bf  = (bf16*)alloc((size_t)L_ * FF_ * C_ * 2);
    float* xproj  = (float*)alloc((size_t)B_ * N_ * C_ * 4);
    float* xw     = (float*)alloc(TC * 4);
    float* pew    = (float*)alloc(TC * 4);
    bf16*  qin    = (bf16*)alloc(TC * 2);
    bf16*  vin    = (bf16*)alloc(TC * 2);
    bf16*  qb     = (bf16*)alloc(TC * 2);
    bf16*  kb     = (bf16*)alloc(TC * 2);
    bf16*  vb     = (bf16*)alloc(TC * 2);
    bf16*  ob     = (bf16*)alloc(TC * 2);
    bf16*  h2b    = (bf16*)alloc(TC * 2);
    bf16*  mid    = (bf16*)alloc((size_t)T_ * FF_ * 2);

    auto cvt = [&](const float* s, bf16* d, int n) {
        k_cvt_bf16<<<(n + 255) / 256, 256, 0, stream>>>(s, d, n);
    };
    cvt(in_x, inp_bf, B_ * N_ * C_);
    cvt(ipw, ipw_bf, C_ * C_);
    cvt(wq, wq_bf, L_ * C_ * C_);
    cvt(wk, wk_bf, L_ * C_ * C_);
    cvt(wv, wv_bf, L_ * C_ * C_);
    cvt(wo, wo_bf, L_ * C_ * C_);
    cvt(w1, w1_bf, L_ * C_ * FF_);
    cvt(w2, w2_bf, L_ * FF_ * C_);

    // input projection: xproj[B*N, C] = input @ ipw + ipb   (M=18432,N=256,K=256)
    {
        dim3 g(C_ / 64, (B_ * N_) / 128);
        k_gemm<0><<<g, 256, 0, stream>>>(inp_bf, ipw_bf, ipb, xproj, nullptr, nullptr,
                                         B_ * N_, C_, C_);
    }
    // bilinear sample + PE -> xw, pew  [T,C]
    k_sample_pe<<<T_, C_, 0, stream>>>(xproj, sphere, ppw, ppb, xw, pew);

    dim3 gC(C_ / 64, T_ / 128);     // (4, 1296)
    dim3 gF(FF_ / 64, T_ / 128);    // (16, 1296)
    const int lnBlocks = T_ / 8;
    const long attnTotal = (long)B_ * N_ * NH_ * 9;
    const int attnBlocks = (int)((attnTotal + 255) / 256);

    for (int l = 0; l < L_; ++l) {
        const size_t wOff = (size_t)l * C_ * C_;
        // LN1 -> qin (hh+pew), vin (hh)
        k_ln<0><<<lnBlocks, 256, 0, stream>>>(xw, pew, ln1g + l * C_, ln1b + l * C_, qin, vin);
        // Q, K, V projections (bf16 out)
        k_gemm<3><<<gC, 256, 0, stream>>>(qin, wq_bf + wOff, bq + l * C_, nullptr, qb, nullptr, T_, C_, C_);
        k_gemm<3><<<gC, 256, 0, stream>>>(qin, wk_bf + wOff, bk + l * C_, nullptr, kb, nullptr, T_, C_, C_);
        k_gemm<3><<<gC, 256, 0, stream>>>(vin, wv_bf + wOff, bv + l * C_, nullptr, vb, nullptr, T_, C_, C_);
        // windowed attention
        k_attn<<<attnBlocks, 256, 0, stream>>>(qb, kb, vb, ob);
        // xw += ob @ wo + bo
        k_gemm<1><<<gC, 256, 0, stream>>>(ob, wo_bf + wOff, bo + l * C_, xw, nullptr, xw, T_, C_, C_);
        // LN2 -> h2
        k_ln<1><<<lnBlocks, 256, 0, stream>>>(xw, nullptr, ln2g + l * C_, ln2b + l * C_, h2b, nullptr);
        // mid = gelu(h2 @ w1 + b1)  (bf16)
        k_gemm<2><<<gF, 256, 0, stream>>>(h2b, w1_bf + (size_t)l * C_ * FF_, b1 + l * FF_,
                                          nullptr, mid, nullptr, T_, FF_, C_);
        // xw += mid @ w2 + b2
        k_gemm<1><<<gC, 256, 0, stream>>>(mid, w2_bf + (size_t)l * FF_ * C_, b2 + l * C_,
                                          xw, nullptr, xw, T_, C_, FF_);
    }

    // reorder windows back to [B, 3H, 3W, C] flat output
    k_reorder<<<(int)(TC / 256), 256, 0, stream>>>(xw, out);
}
